// ContrastiveLoss_45165876084851
// MI455X (gfx1250) — matrix operand order repaired and checked by reading, not compiled, not run on hardware
//
#include <hip/hip_runtime.h>
#include <hip/hip_bf16.h>

typedef __attribute__((ext_vector_type(16))) _Float16 v16h;
typedef __attribute__((ext_vector_type(8)))  _Float16 v8h;
typedef __attribute__((ext_vector_type(8)))  float    v8f;

#define NROWS 8192
#define DIM   256
#define HALFB 4096                 // NROWS/2, positive-pair offset
#define NTILE (NROWS / 16)         // 512 column tiles of 16
#define COLSPLIT 4
#define TPS (NTILE / COLSPLIT)     // 128 tiles per split
#define ROWS_PER_BLK 128           // 8 waves x 16 rows
#define BTILE_BYTES 8192u          // 16 cols x 256 K x 2B

#define LOG2E_F 1.44269504f
#define LN2_F   0.69314718f

// ---------------------------------------------------------------------------
// CDNA5 async global->LDS copy (ASYNCcnt) helpers
// ---------------------------------------------------------------------------
__device__ __forceinline__ void async_ld_b128(unsigned lds_byte_off, const void* g) {
  unsigned long long ga = (unsigned long long)g;
  asm volatile("global_load_async_to_lds_b128 %0, %1, off"
               :: "v"(lds_byte_off), "v"(ga) : "memory");
}
__device__ __forceinline__ void wait_async0() {
  asm volatile("s_wait_asynccnt 0x0" ::: "memory");
}

// ---------------------------------------------------------------------------
// Kernel 1: row-normalize f32 -> f16 copy in workspace. One wave32 per row.
// ---------------------------------------------------------------------------
__global__ __launch_bounds__(256) void nrm_f16_k(const float* __restrict__ feat,
                                                 _Float16* __restrict__ fh) {
  const int lane = threadIdx.x & 31;
  const int row  = (blockIdx.x * blockDim.x + threadIdx.x) >> 5;
  const float* r = feat + (size_t)row * DIM;

  float v[8];
  float s = 0.0f;
#pragma unroll
  for (int i = 0; i < 8; ++i) { v[i] = r[lane + 32 * i]; s += v[i] * v[i]; }
#pragma unroll
  for (int off = 16; off >= 1; off >>= 1) s += __shfl_xor(s, off, 32);

  const float inv = rsqrtf(fmaxf(s, 1e-24f));   // 1/max(||x||,1e-12)

  _Float16* o = fh + (size_t)row * DIM;
#pragma unroll
  for (int i = 0; i < 8; ++i) o[lane + 32 * i] = (_Float16)(v[i] * inv);
}

// ---------------------------------------------------------------------------
// Kernel 2: fused sim GEMM + sum(exp) + positive capture.
// Block = 128 rows (8 waves x 16) x one of 4 column splits.
// B tiles staged in double-buffered LDS via async loads, shared by all waves.
// A is pre-scaled by log2(e) -> cacc = sim*log2e; epilogue is a bare
// v_exp_f32 per element (args in [-1.45,1.45], no range fixup needed).
// |sim| <= 1 (unit rows) -> plain sum of exp is overflow-safe, no max pass.
// ---------------------------------------------------------------------------
__global__ __launch_bounds__(256) void simclr_lse_k(const _Float16* __restrict__ fh,
                                                    float* __restrict__ sePart,
                                                    float* __restrict__ posPart) {
  extern __shared__ char smem[];               // 2 x 8KB B-tile buffers (dyn LDS @0)

  const int lane   = threadIdx.x & 31;
  const int waveId = threadIdx.x >> 5;         // 0..7
  const int half   = lane >> 4;                // 0 | 1
  const int l16    = lane & 15;

  const int rowBlk = blockIdx.x >> 2;          // 0..63
  const int split  = blockIdx.x & 3;           // 0..3
  const int rowBase = rowBlk * ROWS_PER_BLK + waveId * 16;
  const int posBase = (rowBase < HALFB) ? rowBase + HALFB : rowBase - HALFB;
  const int tBase   = split * TPS;
  // Tile indices of the (wave-uniform) diagonal and positive-pair tiles,
  // forced into SGPRs so the per-tile tests compile to scalar branches.
  const int diagT = __builtin_amdgcn_readfirstlane(rowBase >> 4);
  const int posT  = __builtin_amdgcn_readfirstlane(posBase >> 4);

  // ---- A tile: this wave's 16 rows x K=256 in VGPRs (8 x v16h), x log2e.
  // 16-bit A 16x32 layout (ISA 7.12.2):
  //   lanes 0-15  (M=l16): elems 0..7 = K 32c+0..7,  8..15 = K 32c+16..23
  //   lanes 16-31 (M=l16): elems 0..7 = K 32c+8..15, 8..15 = K 32c+24..31
  const _Float16* arow = fh + (size_t)(rowBase + l16) * DIM;
  const _Float16 s2e = (_Float16)LOG2E_F;
  v16h a[8];
#pragma unroll
  for (int c = 0; c < 8; ++c) {
    const v8h lo = *(const v8h*)(arow + 32 * c + half * 8);
    const v8h hi = *(const v8h*)(arow + 32 * c + 16 + half * 8);
#pragma unroll
    for (int i = 0; i < 8; ++i) { a[c][i] = lo[i] * s2e; a[c][i + 8] = hi[i] * s2e; }
  }

  float se[8], pv[8];
#pragma unroll
  for (int m = 0; m < 8; ++m) { se[m] = 0.0f; pv[m] = 0.0f; }

  // ---- prologue stage: tile tBase -> buf 0 (32B per thread, 2 async b128)
  {
    const unsigned lds0 = threadIdx.x * 32u;
    const char* g = (const char*)fh + (size_t)tBase * 16 * DIM * 2 + threadIdx.x * 32;
    async_ld_b128(lds0, g);
    async_ld_b128(lds0 + 16, g + 16);
  }

  for (int i = 0; i < TPS; ++i) {
    const int t   = tBase + i;
    const int buf = i & 1;

    wait_async0();                 // my wave's stage(i) writes landed in LDS
    __syncthreads();               // everyone's landed; prev compute done too

    if (i + 1 < TPS) {             // stage(i+1) into the other buffer
      const unsigned lds0 = (unsigned)(buf ^ 1) * BTILE_BYTES + threadIdx.x * 32u;
      const char* g = (const char*)fh + (size_t)(t + 1) * 16 * DIM * 2 + threadIdx.x * 32;
      async_ld_b128(lds0, g);
      async_ld_b128(lds0 + 16, g + 16);
    }

    // ---- B from LDS: lane holds column l16; elems 0..7 = K 32c+8*half+0..7,
    // elems 8..15 = K 32c+16+8*half+0..7 (two 16B reads per chunk per lane)
    const char* bbase = smem + buf * BTILE_BYTES + l16 * 512 + half * 16;
    v16h bv[8];
#pragma unroll
    for (int c = 0; c < 8; ++c) {
      const v8h blo = *(const v8h*)(bbase + 64 * c);
      const v8h bhi = *(const v8h*)(bbase + 64 * c + 32);
#pragma unroll
      for (int j = 0; j < 8; ++j) { bv[c][j] = blo[j]; bv[c][j + 8] = bhi[j]; }
    }

    v8f cacc = {};
#pragma unroll
    for (int c = 0; c < 8; ++c)
      cacc = __builtin_amdgcn_wmma_f32_16x16x32_f16(
          false, a[c], false, bv[c], (short)0, cacc, false, false);

    // ---- epilogue (cacc = sim * log2e). Diagonal / positive tiles are
    // scalar-branch special cases: fast path for at least 127 of 128 tiles.
    float ex[8];
#pragma unroll
    for (int m = 0; m < 8; ++m)
      ex[m] = __builtin_amdgcn_exp2f(cacc[m]);  // bare v_exp_f32 = exp(sim)

    if (t == diagT) {              // diagonal tile: drop j == i
#pragma unroll
      for (int m = 0; m < 8; ++m)
        se[m] += (l16 == m + 8 * half) ? 0.0f : ex[m];
    } else {
#pragma unroll
      for (int m = 0; m < 8; ++m) se[m] += ex[m];
    }
    if (t == posT) {               // positive-pair tile: capture scaled sim
#pragma unroll
      for (int m = 0; m < 8; ++m)
        pv[m] += (l16 == m + 8 * half) ? cacc[m] : 0.0f;
    }

    __syncthreads();               // done reading buf before stage(i+2) overwrites
  }

  // ---- reduce across the 16 lanes of each half-wave (xor<16 stays in-half)
#pragma unroll
  for (int m = 0; m < 8; ++m) {
#pragma unroll
    for (int off = 1; off < 16; off <<= 1) {
      se[m] += __shfl_xor(se[m], off, 32);
      pv[m] += __shfl_xor(pv[m], off, 32);
    }
  }

  if (l16 == 0) {                  // lanes 0 and 16 each own 8 rows
#pragma unroll
    for (int m = 0; m < 8; ++m) {
      const int row = rowBase + m + 8 * half;
      sePart[row * COLSPLIT + split]  = se[m];
      posPart[row * COLSPLIT + split] = pv[m] * LN2_F;   // unscale: sim = cacc*ln2
    }
  }
}

// ---------------------------------------------------------------------------
// Kernel 3: deterministic ordered reduction -> mean(log(sumexp) - pos).
// ---------------------------------------------------------------------------
__global__ __launch_bounds__(256) void final_k(const float* __restrict__ sePart,
                                               const float* __restrict__ posPart,
                                               float* __restrict__ out) {
  __shared__ float part[256];
  const int tid = threadIdx.x;
  float s = 0.0f;
  for (int r = tid * 32; r < tid * 32 + 32; ++r) {
    float tot = 0.0f, p = 0.0f;
#pragma unroll
    for (int k = 0; k < COLSPLIT; ++k) {
      tot += sePart[r * COLSPLIT + k];
      p   += posPart[r * COLSPLIT + k];
    }
    // ln(tot) = log2(tot) * ln2 ; tot in [~3e3, ~2.3e4] -> no range fixup
    s += __builtin_amdgcn_logf(tot) * LN2_F - p;
  }
  part[tid] = s;
  __syncthreads();
  if (tid == 0) {
    float tot = 0.0f;
    for (int i = 0; i < 256; ++i) tot += part[i];
    out[0] = tot / (float)NROWS;
  }
}

extern "C" void kernel_launch(void* const* d_in, const int* in_sizes, int n_in,
                              void* d_out, int out_size, void* d_ws, size_t ws_size,
                              hipStream_t stream) {
  (void)in_sizes; (void)n_in; (void)out_size; (void)ws_size;
  const float* feat = (const float*)d_in[0];
  _Float16* fh    = (_Float16*)d_ws;                                        // 4 MB
  float* sePart   = (float*)((char*)d_ws + (size_t)NROWS * DIM * 2);        // 128 KB
  float* posPart  = sePart + (size_t)NROWS * COLSPLIT;                      // 128 KB

  nrm_f16_k<<<NROWS / 8, 256, 0, stream>>>(feat, fh);
  simclr_lse_k<<<(NROWS / ROWS_PER_BLK) * COLSPLIT, 256, 2 * BTILE_BYTES, stream>>>(
      fh, sePart, posPart);
  final_k<<<1, 256, 0, stream>>>(sePart, posPart, (float*)d_out);
}
